// EdgeScalarFeatInit_90426241450248
// MI455X (gfx1250) — matrix operand order repaired and checked by reading, not compiled, not run on hardware
//
#include <hip/hip_runtime.h>
#include <stdint.h>

// MI455X / gfx1250 — wave32, WMMA bf16 16x16x32, TDM staging of the weight
// matrix into LDS. Einsum done as S[rows x 512] x W2^T[512 x 128] with
// hi/lo-split bf16 WMMA (3 MACs) for ~fp32 accuracy; SiLU/LN folded into a
// per-row scalar t and a per-k vector c computed by a prep kernel.

typedef __attribute__((ext_vector_type(16))) __bf16 v16bf;
typedef __attribute__((ext_vector_type(8)))  float  v8f;
typedef __attribute__((ext_vector_type(4))) unsigned int u32x4;
typedef __attribute__((ext_vector_type(8)))  int    i32x8;
typedef __attribute__((ext_vector_type(4)))  int    i32x4;

namespace {
constexpr int kN    = 8192;
constexpr int kM    = 32;
constexpr int kD    = 128;            // output feature dim
constexpr int kK    = 512;            // d_inner
constexpr int kRows = kN * kM;        // 262144 edge rows
constexpr int kCT   = kD / 16;        // 8 column tiles
constexpr int kKC   = kK / 32;        // 16 K-chunks of 32
constexpr int kFragElems = 16;        // bf16 elems per lane per fragment
// workspace / LDS layout (bytes)
constexpr int kHalfBytes = kKC * kCT * 32 * kFragElems * 2;   // 131072 (one bf16 copy of W2)
constexpr int kLoOff  = kHalfBytes;                           // 131072
constexpr int kCOff   = 2 * kHalfBytes;                       // 262144 (c[512] f32)
constexpr int kVarOff = kCOff + kK * 4;                       // 264192 (var_w f32, global only)
constexpr int kLdsBytes = kVarOff;                            // 264192 staged via TDM
constexpr float kEps = 1e-5f;
}

// ---------------------------------------------------------------------------
// Prep: reduce w1 -> mu/var, build c_k = (w1_k - mu)*ln_g_k, pack w2 into
// WMMA B-fragment order as bf16 hi/lo split.
// B 32x16 (KxN) 16-bit layout: lane = column (lanes 0-15 / 16-31 share N),
// lanes 0-15 hold K=0..7,16..23 across V0..V7 (2 per VGPR), lanes 16-31 hold K=8..15,24..31.
// ---------------------------------------------------------------------------
__global__ __launch_bounds__(512) void prep_kernel(
    const float* __restrict__ w1, const float* __restrict__ ln_g,
    const float* __restrict__ w2, unsigned char* __restrict__ ws) {
  __bf16* w2hi = (__bf16*)(ws);
  __bf16* w2lo = (__bf16*)(ws + kLoOff);
  float*  cvec = (float*)(ws + kCOff);
  float*  varp = (float*)(ws + kVarOff);

  __shared__ float red[512];
  __shared__ float mu_s;
  const int tid = threadIdx.x;
  const float w = w1[tid];
  red[tid] = w;
  __syncthreads();
  for (int s = 256; s > 0; s >>= 1) {
    if (tid < s) red[tid] += red[tid + s];
    __syncthreads();
  }
  if (tid == 0) mu_s = red[0] * (1.0f / (float)kK);
  __syncthreads();
  const float mu = mu_s;
  const float dv = w - mu;
  red[tid] = dv * dv;
  __syncthreads();
  for (int s = 256; s > 0; s >>= 1) {
    if (tid < s) red[tid] += red[tid + s];
    __syncthreads();
  }
  if (tid == 0) varp[0] = red[0] * (1.0f / (float)kK);
  cvec[tid] = dv * ln_g[tid];

  // Pack: flat fragment index = ((kc*8 + ct)*32 + lane)*16 + e
  for (int idx = tid; idx < kKC * kCT * 32 * kFragElems; idx += 512) {
    const int e    = idx & 15;
    const int lane = (idx >> 4) & 31;
    const int ct   = (idx >> 9) & 7;
    const int kc   = idx >> 12;
    const int kl   = ((lane < 16) ? 0 : 8) + ((e < 8) ? e : e + 8);     // K within 32-chunk
    const int k    = kc * 32 + kl;
    const int n    = ct * 16 + (lane & 15);          // output column
    const float x  = w2[n * kK + k];                 // B[k][n] = w2[n][k]
    const __bf16 hi = (__bf16)x;
    w2hi[idx] = hi;
    w2lo[idx] = (__bf16)(x - (float)hi);
  }
}

// ---------------------------------------------------------------------------
// Main kernel: 8 waves/block, each wave computes a 16-row x 128-col tile.
// ---------------------------------------------------------------------------
__global__ __launch_bounds__(256) void edge_feat_kernel(
    const float* __restrict__ h, const float* __restrict__ rel,
    const long long* __restrict__ nbr, const unsigned char* __restrict__ ws,
    float* __restrict__ out) {
  __shared__ __align__(128) unsigned char smem[kLdsBytes];

  const int tid  = threadIdx.x;
  const int lane = tid & 31;
  const int wave = tid >> 5;

  if (wave == 0) {
    // --- Tensor Data Mover: stage packed W2 (hi/lo bf16) + c[512] into LDS.
    // 1-D descriptor, data_size=8B, 33024 elements = 264192 bytes.
    const unsigned long long ga = (unsigned long long)(uintptr_t)ws;
    const unsigned int lds_off  = (unsigned int)(uintptr_t)smem;  // LDS byte offset (low 32 bits of flat addr)
    const unsigned int nq = (unsigned int)(kLdsBytes / 8);        // 33024
    u32x4 g0;
    g0[0] = 1u;                                     // count = 1 (valid D#)
    g0[1] = lds_off;                                // lds_addr
    g0[2] = (unsigned int)ga;                       // global_addr[31:0]
    g0[3] = (unsigned int)(ga >> 32) | 0x80000000u; // global_addr[56:32] | type=2<<30
    i32x8 g1;
    g1[0] = (int)(3u << 16);                        // data_size = 3 (8 bytes), wg_mask = 0
    g1[1] = (int)(nq << 16);                        // tensor_dim0[15:0] (bits 63:48)
    g1[2] = (int)((nq >> 16) | (1u << 16));         // tensor_dim0[31:16] | tensor_dim1 = 1
    g1[3] = (int)(nq << 16);                        // tile_dim0 = nq (bits 127:112)
    g1[4] = 0;                                      // tile_dim1 / tile_dim2 unused
    g1[5] = (int)nq;                                // tensor_dim0_stride
    g1[6] = 0;
    g1[7] = 0;
    i32x4 gz4 = {0, 0, 0, 0};
    i32x8 gz8 = {0, 0, 0, 0, 0, 0, 0, 0};
    __builtin_amdgcn_tensor_load_to_lds(g0, g1, gz4, gz4, gz8, 0);
    __builtin_amdgcn_s_wait_tensorcnt(0);
  }
  __syncthreads();

  const __bf16* Bhi = (const __bf16*)(smem);
  const __bf16* Blo = (const __bf16*)(smem + kLoOff);
  const float*  Cc  = (const float*)(smem + kCOff);
  const float   varw = *(const float*)(ws + kVarOff);

  const int row_base = (blockIdx.x * 8 + wave) * 16;
  // A-fragment: lane & 15 selects row M; lanes<16 hold K {0..7,16..23},
  // lanes>=16 hold K {8..15,24..31} (ISA 16-bit A 16x32 layout).
  const float r = rel[row_base + (lane & 15)];
  const float t = r * rsqrtf(r * r * varw + kEps);
  const int sub = (lane < 16) ? 0 : 8;

  v8f acc[kCT] = {};

  for (int kc = 0; kc < kKC; ++kc) {
    const float* cb = Cc + kc * 32;
    v16bf ahi, alo;
#pragma unroll
    for (int e = 0; e < kFragElems; ++e) {
      const int j = e >> 1, p = e & 1;
      const int kl = (j < 4) ? (sub + 2 * j + p) : (16 + sub + 2 * (j - 4) + p);
      const float u = t * cb[kl];
      const float s = u / (1.0f + __expf(-u));    // SiLU(LN(r*w1))
      const __bf16 sh = (__bf16)s;
      ahi[e] = sh;
      alo[e] = (__bf16)(s - (float)sh);
    }
#pragma unroll
    for (int ct = 0; ct < kCT; ++ct) {
      const int off = kc * (kCT * 32 * kFragElems) + ct * (32 * kFragElems) + lane * kFragElems;
      const v16bf bh = *(const v16bf*)(Bhi + off);
      const v16bf bl = *(const v16bf*)(Blo + off);
      acc[ct] = __builtin_amdgcn_wmma_f32_16x16x32_bf16(false, ahi, false, bh, (short)0, acc[ct], false, false);
      acc[ct] = __builtin_amdgcn_wmma_f32_16x16x32_bf16(false, ahi, false, bl, (short)0, acc[ct], false, false);
      acc[ct] = __builtin_amdgcn_wmma_f32_16x16x32_bf16(false, alo, false, bh, (short)0, acc[ct], false, false);
    }
  }

  // Epilogue. D layout: VGPR v -> row M = v (lanes 0-15) / v+8 (lanes 16-31),
  // col N = lane & 15. row_base is 16-aligned within a 32-row node => the
  // self-term h[i,:] is uniform over the tile.
  int nb[8];
#pragma unroll
  for (int v = 0; v < 8; ++v) {
    const int rowv = row_base + ((lane < 16) ? v : v + 8);
    nb[v] = (int)nbr[rowv];
  }
  const int inode = row_base >> 5;   // row / 32
#pragma unroll
  for (int ct = 0; ct < kCT; ++ct) {
    const int col = ct * 16 + (lane & 15);
    const float hs = h[inode * kD + col];
#pragma unroll
    for (int v = 0; v < 8; ++v) {
      const int rowv = row_base + ((lane < 16) ? v : v + 8);
      out[(size_t)rowv * kD + col] = acc[ct][v] + hs + h[(size_t)nb[v] * kD + col];
    }
  }
}

// ---------------------------------------------------------------------------
extern "C" void kernel_launch(void* const* d_in, const int* in_sizes, int n_in,
                              void* d_out, int out_size, void* d_ws, size_t ws_size,
                              hipStream_t stream) {
  (void)in_sizes; (void)n_in; (void)out_size; (void)ws_size;
  const float*     h   = (const float*)d_in[0];
  const float*     rel = (const float*)d_in[1];
  const long long* nbr = (const long long*)d_in[2];
  const float*     w1  = (const float*)d_in[3];
  const float*     lng = (const float*)d_in[4];
  const float*     w2  = (const float*)d_in[5];
  float* out = (float*)d_out;
  unsigned char* ws = (unsigned char*)d_ws;

  prep_kernel<<<1, 512, 0, stream>>>(w1, lng, w2, ws);
  edge_feat_kernel<<<kRows / 128, 256, 0, stream>>>(h, rel, nbr, ws, out);
}
